// BiModalAttention_76364518523410
// MI455X (gfx1250) — matrix operand order repaired and checked by reading, not compiled, not run on hardware
//
#include <hip/hip_runtime.h>
#include <hip/hip_bf16.h>
#include <math.h>

// ---------------------------------------------------------------------------
// BiModalAttention for MI455X (gfx1250): flash-attention style, bf16 WMMA.
// Transposed formulation: S^T = K Q^T, O^T = V^T P^T  (softmax axis in VGPRs;
// reductions = in-lane VALU tree + one v_permlanex16_b32).
// This round: one-time fp32->bf16 pre-conversion into d_ws (halves loop-body
// bytes, kills per-tile cvt VALU) + GLOBAL_LOAD_ASYNC_TO_LDS_B128 for the sK
// tile copy (ASYNCcnt-tracked DMA, no VGPR round-trip) with s_wait_asynccnt.
// ---------------------------------------------------------------------------

typedef __attribute__((ext_vector_type(16))) __bf16 v16bf;
typedef __attribute__((ext_vector_type(8)))  __bf16 v8bf;
typedef __attribute__((ext_vector_type(8)))  float  v8f;
typedef __attribute__((ext_vector_type(4)))  float  v4f;
typedef __attribute__((ext_vector_type(8)))  int    v8i;

#define B_BATCH 8
#define S_LEN   2048
#define D_DIM   128
#define BM      128         // Q rows per workgroup (8 waves x 16 rows)
#define BN      32          // KV rows per tile (one WMMA K-step for PV)
#define NWAVES  8
#define KT_STRIDE 136       // 128 + 8 bf16 pad: 272B rows -> conflict-free
#define VT_STRIDE 40        // 32 + 8 bf16 pad:  80B rows

static __device__ __forceinline__ void sched_fence() {
#if __has_builtin(__builtin_amdgcn_sched_barrier)
  __builtin_amdgcn_sched_barrier(0);
#endif
}

static __device__ __forceinline__ v16bf cat16(v8bf a, v8bf b) {
  v16bf r;
#pragma unroll
  for (int i = 0; i < 8; ++i) { r[i] = a[i]; r[i + 8] = b[i]; }
  return r;
}

// Exchange with the partner lane in the other 16-lane half (lane ^ 16).
static __device__ __forceinline__ int permx16i(int v) {
#if __has_builtin(__builtin_amdgcn_permlanex16)
  return __builtin_amdgcn_permlanex16(v, v, 0x76543210, 0xFEDCBA98, false, false);
#else
  return __shfl_xor(v, 16, 32);
#endif
}
static __device__ __forceinline__ float permx16f(float v) {
  return __builtin_bit_cast(float, permx16i(__builtin_bit_cast(int, v)));
}

static __device__ __forceinline__ unsigned pack2bf(__bf16 a, __bf16 b) {
  unsigned la = (unsigned)__builtin_bit_cast(unsigned short, a);
  unsigned lb = (unsigned)__builtin_bit_cast(unsigned short, b);
  return la | (lb << 16);
}

// Low 32 bits of a generic pointer into LDS == DS byte address.
static __device__ __forceinline__ unsigned lds_addr(const void* p) {
  return (unsigned)(uintptr_t)p;
}

// ---------------------------------------------------------------------------
// One-time fp32 -> bf16 conversion (writes d_ws); trivially bandwidth-bound.
// ---------------------------------------------------------------------------
__global__ __launch_bounds__(256)
void cvt_to_bf16(const float* __restrict__ src, __bf16* __restrict__ dst) {
  const size_t i = ((size_t)blockIdx.x * 256 + threadIdx.x) * 8;
  v4f a = *(const v4f*)(src + i);
  v4f b = *(const v4f*)(src + i + 4);
  v8bf o;
#pragma unroll
  for (int k = 0; k < 4; ++k) { o[k] = (__bf16)a[k]; o[4 + k] = (__bf16)b[k]; }
  *(v8bf*)(dst + i) = o;
}

// ---------------------------------------------------------------------------
// Attention kernel. BF16SRC=true: read pre-converted bf16 (xb/yb) and use the
// async-to-LDS DMA path. BF16SRC=false: self-converting fp32 fallback.
// ---------------------------------------------------------------------------
template <bool BF16SRC>
__global__ __launch_bounds__(256)
void bimodal_attn_wmma(const float* __restrict__ xf, const float* __restrict__ yf,
                       const __bf16* __restrict__ xb, const __bf16* __restrict__ yb,
                       float* __restrict__ out)
{
  const int b    = blockIdx.y;
  const int pass = blockIdx.z;                 // 0: Q=x,KV=y  1: Q=y,KV=x
  const float*  __restrict__ qptrf = pass ? yf : xf;   // gating (fp32 exact)
  const float*  __restrict__ kptrf = pass ? xf : yf;
  const __bf16* __restrict__ qptrb = pass ? yb : xb;
  const __bf16* __restrict__ kptrb = pass ? xb : yb;

  const int q0     = blockIdx.x * BM;
  const int lane   = threadIdx.x & 31;
  const int wave   = threadIdx.x >> 5;
  const int halfhi = lane >> 4;                // 0: lanes 0-15, 1: lanes 16-31
  const int ln     = lane & 15;

  __shared__ __align__(16) __bf16 sK [BN * KT_STRIDE];      // KV tile [t][d]
  __shared__ __align__(16) __bf16 sVt[D_DIM * VT_STRIDE];   // KV tile [d][t]

  // ---- Q^T fragments (loop-invariant B operand): lane = q row s ----
  const int srow = q0 + wave * 16 + ln;
  v16bf qf[4];
  if constexpr (BF16SRC) {
    const __bf16* qb = qptrb + ((size_t)b * S_LEN + srow) * D_DIM;
#pragma unroll
    for (int c = 0; c < 4; ++c) {
      const int d0 = 32 * c + 16 * halfhi;     // 16 contiguous d per lane
      qf[c] = cat16(*(const v8bf*)(qb + d0), *(const v8bf*)(qb + d0 + 8));
    }
  } else {
    const float* qbase = qptrf + ((size_t)b * S_LEN + srow) * D_DIM;
#pragma unroll
    for (int c = 0; c < 4; ++c) {
      const int d0 = 32 * c + 16 * halfhi;
      v4f f0 = *(const v4f*)(qbase + d0);
      v4f f1 = *(const v4f*)(qbase + d0 + 4);
      v4f f2 = *(const v4f*)(qbase + d0 + 8);
      v4f f3 = *(const v4f*)(qbase + d0 + 12);
      v16bf q;
#pragma unroll
      for (int i = 0; i < 4; ++i) {
        q[i]      = (__bf16)f0[i];
        q[4 + i]  = (__bf16)f1[i];
        q[8 + i]  = (__bf16)f2[i];
        q[12 + i] = (__bf16)f3[i];
      }
      qf[c] = q;
    }
  }

  // ---- O^T accumulator: tile j, lane = s, VGPR r -> d = 16j + 8*halfhi + r
  v8f acc[8];
#pragma unroll
  for (int j = 0; j < 8; ++j)
#pragma unroll
    for (int r = 0; r < 8; ++r) acc[j][r] = 0.0f;
  float mrun = -__builtin_inff();
  float lrun = 0.0f;

  const float*  kbf_g = kptrf + (size_t)b * S_LEN * D_DIM;
  const __bf16* kbb_g = BF16SRC ? (kptrb + (size_t)b * S_LEN * D_DIM) : nullptr;

  for (int kv0 = 0; kv0 < S_LEN; kv0 += BN) {
    __syncthreads();
    // ---- cooperative tile load, dual layout (pair of t-rows x 8 d) ----
    {
      const int tid = threadIdx.x;
      const int tp  = tid >> 4;           // 0..15 -> rows 2tp, 2tp+1
      const int dg  = (tid & 15) * 8;     // d base: 0,8,...,120
      if constexpr (BF16SRC) {
        const __bf16* s0p = kbb_g + (size_t)(kv0 + 2 * tp) * D_DIM + dg;
        const __bf16* s1p = s0p + D_DIM;
        if (kv0 + BN < S_LEN)
          __builtin_prefetch(s0p + (size_t)BN * D_DIM, 0, 0);
        // async DMA the row-major chunks straight into sK (ASYNCcnt)
        const unsigned l0 = lds_addr(&sK[(2 * tp)     * KT_STRIDE + dg]);
        const unsigned l1 = lds_addr(&sK[(2 * tp + 1) * KT_STRIDE + dg]);
        asm volatile("global_load_async_to_lds_b128 %0, %1, off"
                     :: "v"(l0), "v"((unsigned long long)(uintptr_t)s0p)
                     : "memory");
        asm volatile("global_load_async_to_lds_b128 %0, %1, off"
                     :: "v"(l1), "v"((unsigned long long)(uintptr_t)s1p)
                     : "memory");
        // regular loads feed the packed transposed copy (L2-hot re-read)
        v8bf r0 = *(const v8bf*)s0p;
        v8bf r1 = *(const v8bf*)s1p;
#pragma unroll
        for (int i = 0; i < 8; ++i)
          *(unsigned*)&sVt[(dg + i) * VT_STRIDE + 2 * tp] = pack2bf(r0[i], r1[i]);
        asm volatile("s_wait_asynccnt 0x0" ::: "memory");
      } else {
        const float* src0 = kbf_g + (size_t)(kv0 + 2 * tp) * D_DIM + dg;
        const float* src1 = src0 + D_DIM;
        if (kv0 + BN < S_LEN)
          __builtin_prefetch(src0 + (size_t)BN * D_DIM, 0, 0);
        v4f a0 = *(const v4f*)(src0);
        v4f a1 = *(const v4f*)(src0 + 4);
        v4f b0 = *(const v4f*)(src1);
        v4f b1 = *(const v4f*)(src1 + 4);
        v8bf r0, r1;
#pragma unroll
        for (int i = 0; i < 4; ++i) {
          r0[i]     = (__bf16)a0[i];
          r0[4 + i] = (__bf16)a1[i];
          r1[i]     = (__bf16)b0[i];
          r1[4 + i] = (__bf16)b1[i];
        }
        *(v8bf*)&sK[(2 * tp)     * KT_STRIDE + dg] = r0;
        *(v8bf*)&sK[(2 * tp + 1) * KT_STRIDE + dg] = r1;
#pragma unroll
        for (int i = 0; i < 8; ++i)
          *(unsigned*)&sVt[(dg + i) * VT_STRIDE + 2 * tp] = pack2bf(r0[i], r1[i]);
      }
    }
    __syncthreads();

    // ---- preload ALL K-row fragments, pinned ahead of the WMMA chain ----
    v16bf af[8];            // af[c] = t-sub0, af[4+c] = t-sub1
#pragma unroll
    for (int c = 0; c < 4; ++c) {
      const int koff = 32 * c + (halfhi ? 16 : 0);
      af[c]     = cat16(*(const v8bf*)&sK[ln        * KT_STRIDE + koff],
                        *(const v8bf*)&sK[ln        * KT_STRIDE + koff + 8]);
      af[4 + c] = cat16(*(const v8bf*)&sK[(16 + ln) * KT_STRIDE + koff],
                        *(const v8bf*)&sK[(16 + ln) * KT_STRIDE + koff + 8]);
    }
    sched_fence();   // keep the 16 ds_load_b128 batched before the WMMAs

    // ---- S^T = K * Q^T : two 16x16 C tiles (t-subtiles), K-dim = 128 ----
    v8f s0, s1;
#pragma unroll
    for (int r = 0; r < 8; ++r) { s0[r] = 0.0f; s1[r] = 0.0f; }
#pragma unroll
    for (int c = 0; c < 4; ++c) {
      s0 = __builtin_amdgcn_wmma_f32_16x16x32_bf16(false, af[c], false,
             qf[c], (short)0, s0, false, false);
      s1 = __builtin_amdgcn_wmma_f32_16x16x32_bf16(false, af[4 + c], false,
             qf[c], (short)0, s1, false, false);
    }
    // Lane (s, lo) holds t = {r} U {16+r}; partner (s, hi) holds {8+r, 24+r}.

    // ---- V^T fragments: issued here so softmax VALU hides their latency ----
    v16bf vf[8];
#pragma unroll
    for (int j = 0; j < 8; ++j) {
      const int drow = 16 * j + ln;
      const int kb   = halfhi ? 16 : 0;
      vf[j] = cat16(*(const v8bf*)&sVt[drow * VT_STRIDE + kb],
                    *(const v8bf*)&sVt[drow * VT_STRIDE + kb + 8]);
    }

    // ---- online softmax (per-lane scalar state) ----
    float sc;
    {
      float mx = fmaxf(s0[0], s1[0]);
#pragma unroll
      for (int r = 1; r < 8; ++r) mx = fmaxf(mx, fmaxf(s0[r], s1[r]));
      mx = fmaxf(mx, permx16f(mx));
      const float mn = fmaxf(mrun, mx);
      sc = __expf(mrun - mn);
      float ps = 0.0f;
#pragma unroll
      for (int r = 0; r < 8; ++r) {
        s0[r] = __expf(s0[r] - mn);
        s1[r] = __expf(s1[r] - mn);
        ps += s0[r] + s1[r];
      }
      lrun = lrun * sc + (ps + permx16f(ps));
      mrun = mn;
    }

    // ---- build P^T B-fragment in registers (no LDS round-trip) ----
    v16bf pfrag;
    {
      v16bf p01;
#pragma unroll
      for (int r = 0; r < 8; ++r) {
        p01[r]     = (__bf16)s0[r];   // tile0: t = r + 8*halfhi
        p01[8 + r] = (__bf16)s1[r];   // tile1: t = 16 + r + 8*halfhi
      }
      v8i pi = __builtin_bit_cast(v8i, p01);
      v8i po;
#pragma unroll
      for (int v = 0; v < 4; ++v) {
        const int sw0 = permx16i(pi[v]);       // partner's tile0 pairs
        const int sw1 = permx16i(pi[4 + v]);   // partner's tile1 pairs
        po[v]     = halfhi ? sw1 : pi[v];      // t kb+0..7
        po[4 + v] = halfhi ? pi[4 + v] : sw0;  // t kb+8..15
      }
      pfrag = __builtin_bit_cast(v16bf, po);
    }

    // ---- O^T += V^T * P^T, rescale interleaved (co-executes with WMMA) ----
#pragma unroll
    for (int j = 0; j < 8; ++j) {
      v8f a = acc[j];
#pragma unroll
      for (int r = 0; r < 8; ++r) a[r] *= sc;
      acc[j] = __builtin_amdgcn_wmma_f32_16x16x32_bf16(false, vf[j], false,
                 pfrag, (short)0, a, false, false);
    }
  }

  // ---- epilogue: normalize, gate by Q operand (fp32), b128 stores ----
  {
    const float inv = 1.0f / lrun;
    const float* grow = qptrf + ((size_t)b * S_LEN + srow) * D_DIM;
    float* orow = out + ((size_t)b * S_LEN + srow) * (2 * D_DIM)
                      + (size_t)pass * D_DIM;
#pragma unroll
    for (int j = 0; j < 8; ++j) {
      const int d0 = 16 * j + 8 * halfhi;     // 8 consecutive d per lane
      v4f g0 = *(const v4f*)(grow + d0);
      v4f g1 = *(const v4f*)(grow + d0 + 4);
      v4f o0, o1;
#pragma unroll
      for (int i = 0; i < 4; ++i) {
        o0[i] = acc[j][i]     * inv * g0[i];
        o1[i] = acc[j][4 + i] * inv * g1[i];
      }
      *(v4f*)(orow + d0)     = o0;
      *(v4f*)(orow + d0 + 4) = o1;
    }
  }
}

extern "C" void kernel_launch(void* const* d_in, const int* in_sizes, int n_in,
                              void* d_out, int out_size, void* d_ws, size_t ws_size,
                              hipStream_t stream) {
  (void)in_sizes; (void)n_in; (void)out_size;
  const float* x = (const float*)d_in[0];
  const float* y = (const float*)d_in[1];
  float* out = (float*)d_out;

  const size_t nelem = (size_t)B_BATCH * S_LEN * D_DIM;      // per tensor
  const size_t need  = 2 * nelem * sizeof(__bf16);           // ~8.4 MB
  dim3 grid(S_LEN / BM, B_BATCH, 2);
  dim3 block(256);

  if (d_ws != nullptr && ws_size >= need) {
    __bf16* xb = (__bf16*)d_ws;
    __bf16* yb = xb + nelem;
    const int cvt_blocks = (int)(nelem / (256 * 8));         // 1024
    cvt_to_bf16<<<cvt_blocks, 256, 0, stream>>>(x, xb);
    cvt_to_bf16<<<cvt_blocks, 256, 0, stream>>>(y, yb);
    bimodal_attn_wmma<true><<<grid, block, 0, stream>>>(x, y, xb, yb, out);
  } else {
    bimodal_attn_wmma<false><<<grid, block, 0, stream>>>(x, y, nullptr, nullptr, out);
  }
}
